// EleAttG_80633716015531
// MI455X (gfx1250) — matrix operand order repaired and checked by reading, not compiled, not run on hardware
//
#include <hip/hip_runtime.h>
#include <hip/hip_bf16.h>

// ---------------------------------------------------------------------------
// EleAttG-GRU fused: B=8192, I=H=1024.
//   phase0: fp32 -> bf16 conversion of x, hx, W_xa, W_ha, W_ih, W_hh
//   phase1: xg = sigmoid(x W_xa^T + hx W_ha^T + b_xa + b_ha) * x   (bf16 out)
//   phase2: GRU cell via 12 fused WMMA accumulators -> hy (fp32)
// Matmuls: v_wmma_f32_16x16x32_bf16 (wave32, fp32 accumulate).
// Activation tiles staged into LDS by the Tensor Data Mover (double-buffered,
// s_wait_tensorcnt), with a register-staged fallback if the builtin is absent.
// ---------------------------------------------------------------------------

#define BDIM  1024
#define BATCH 8192
#define LDSP  40          // padded LDS row pitch in bf16 elems (80 B, bank-friendly)

typedef __attribute__((ext_vector_type(16))) __bf16 v16bf;
typedef __attribute__((ext_vector_type(8)))  __bf16 v8bf;
typedef __attribute__((ext_vector_type(8)))  float  v8f;
typedef __attribute__((ext_vector_type(4)))  unsigned int u32x4;
typedef __attribute__((ext_vector_type(8)))  int i32x8;
typedef __attribute__((ext_vector_type(4)))  int i32x4;

#if defined(__has_builtin)
#if __has_builtin(__builtin_amdgcn_tensor_load_to_lds)
#define USE_TDM 1
#endif
#endif
#ifndef USE_TDM
#define USE_TDM 0
#endif

union Frag { v16bf v; uint4 u[2]; };
union Acc  { v8f  v; float f[8]; };

__device__ __forceinline__ float sigmoid_f(float x) {
  return 1.0f / (1.0f + __expf(-x));
}
__device__ __forceinline__ float tanh_f(float x) {
  float e = __expf(-2.0f * x);
  return (1.0f - e) / (1.0f + e);
}

// A fragment (16x32 bf16, MxK) from an LDS tile with row pitch LDSP elems.
__device__ __forceinline__ void loadA(Frag& f, const __bf16* s, int lane) {
  const int row = lane & 15;
  const int kb  = (lane >> 4) << 3;                 // 0 or 8
  f.u[0] = *(const uint4*)(s + row * LDSP + kb);
  f.u[1] = *(const uint4*)(s + row * LDSP + kb + 16);
}

// B fragment (32x16 bf16, KxN) direct from global weight W[Nrows, 1024] row-major.
__device__ __forceinline__ void loadB(Frag& f, const __bf16* __restrict__ W,
                                      int ncol, int k0, int lane) {
  const int n  = ncol + (lane & 15);
  const int kb = k0 + ((lane >> 4) << 3);           // 0 or 8
  const uint4* p = (const uint4*)(W + (size_t)n * BDIM + kb);
  f.u[0] = p[0];
  f.u[1] = p[2];
}

#if USE_TDM
// Issue one TDM 2D tile load: rows x 32 bf16 from a [8192 x 1024] bf16 tensor
// into LDS with an 80-byte row pitch (64B data + 16B pad via pad fields).
__device__ __forceinline__ void tdm_load_tile(unsigned lds_off, const __bf16* gsrc,
                                              int rows) {
  const unsigned long long ga = (unsigned long long)(size_t)(const void*)gsrc;
  u32x4 g0;
  g0[0] = 1u;                                        // count=1 (valid descriptor)
  g0[1] = lds_off;                                   // lds_addr (bytes)
  g0[2] = (unsigned)(ga & 0xFFFFFFFFu);              // global_addr[31:0]
  g0[3] = (unsigned)((ga >> 32) & 0x01FFFFFFu)       // global_addr[56:32]
        | (2u << 30);                                // type = 2 ("image")
  i32x8 g1;
  g1[0] = (int)((1u << 16)     // data_size = 1 -> 2 bytes
              | (1u << 20)     // pad_enable
              | (3u << 22)     // pad_interval: 16 DWORDs (= 64B row)
              | (3u << 25));   // pad_amount:   4 DWORDs (= 16B pad)
  g1[1] = (int)(1024u << 16);  // tensor_dim0 = 1024 (low 16 -> bits[79:64..])
  g1[2] = (int)(8192u << 16);  // tensor_dim0 hi=0 | tensor_dim1 = 8192 (low 16)
  g1[3] = (int)(32u << 16);    // tensor_dim1 hi=0 | tile_dim0 = 32
  g1[4] = rows;                // tile_dim1 = rows, tile_dim2 = 0
  g1[5] = 1024;                // tensor_dim0_stride (elements), low 32
  g1[6] = 0;                   // stride hi | tensor_dim1_stride (unused, 2D tile)
  g1[7] = 0;
  const i32x4 z = {0, 0, 0, 0};
#if __clang_major__ >= 23
  const i32x8 z8 = {0, 0, 0, 0, 0, 0, 0, 0};
  __builtin_amdgcn_tensor_load_to_lds(g0, g1, z, z, z8, 0);
#else
  __builtin_amdgcn_tensor_load_to_lds(g0, g1, z, z, 0);
#endif
}
#endif

// ---------------------------------------------------------------------------
__global__ __launch_bounds__(256) void cvt_bf16_kernel(const float* __restrict__ in,
                                                       __bf16* __restrict__ out, int n) {
  int i = (blockIdx.x * 256 + threadIdx.x) * 8;
  if (i + 8 <= n) {
    float4 a = *(const float4*)(in + i);
    float4 b = *(const float4*)(in + i + 4);
    v8bf o;
    o[0] = (__bf16)a.x; o[1] = (__bf16)a.y; o[2] = (__bf16)a.z; o[3] = (__bf16)a.w;
    o[4] = (__bf16)b.x; o[5] = (__bf16)b.y; o[6] = (__bf16)b.z; o[7] = (__bf16)b.w;
    *(v8bf*)(out + i) = o;
  }
}

// ---------------------------------------------------------------------------
// Phase 1: workgroup tile 128(M) x 64(N); 8 waves; wave = (mq, nt);
// each wave owns 4 M-tiles for register-level reuse of B fragments.
__global__ __launch_bounds__(256) void phase1_kernel(
    const __bf16* __restrict__ xb,  const __bf16* __restrict__ hb,
    const __bf16* __restrict__ wxa, const __bf16* __restrict__ wha,
    const float*  __restrict__ bxa, const float*  __restrict__ bha,
    const float*  __restrict__ x,   __bf16* __restrict__ xg)
{
  __shared__ __bf16 sx[2 * 128 * LDSP];
  __shared__ __bf16 sh[2 * 128 * LDSP];
  const int tid  = threadIdx.x;
  const int wave = tid >> 5;
  const int lane = tid & 31;
  const int mq   = wave >> 2;                 // 0..1
  const int nt   = wave & 3;                  // 0..3
  const int mBlk = blockIdx.y * 128;
  const int nBlk = blockIdx.x * 64;
  const int ncol = nBlk + nt * 16;
  const int BUFE = 128 * LDSP;                // elems per buffer

  Acc acc[4];
#pragma unroll
  for (int j = 0; j < 4; ++j)
#pragma unroll
    for (int v = 0; v < 8; ++v) acc[j].f[v] = 0.0f;

#if USE_TDM
  const unsigned sx_lds = (unsigned)(size_t)(void*)&sx[0];
  const unsigned sh_lds = (unsigned)(size_t)(void*)&sh[0];
  if (wave == 0) {
    tdm_load_tile(sx_lds, xb + (size_t)mBlk * BDIM, 128);
    tdm_load_tile(sh_lds, hb + (size_t)mBlk * BDIM, 128);
  }
#else
  const int r    = tid >> 1;                  // 0..127
  const int half = tid & 1;                   // 32B half of a 64B K-row
#endif

  for (int k0 = 0; k0 < BDIM; k0 += 32) {
    const int cur = (k0 >> 5) & 1;
#if USE_TDM
    if (wave == 0) {
      if (k0 + 32 < BDIM) {
        const int nxt = cur ^ 1;
        tdm_load_tile(sx_lds + (unsigned)(nxt * BUFE * 2),
                      xb + (size_t)mBlk * BDIM + k0 + 32, 128);
        tdm_load_tile(sh_lds + (unsigned)(nxt * BUFE * 2),
                      hb + (size_t)mBlk * BDIM + k0 + 32, 128);
        __builtin_amdgcn_s_wait_tensorcnt(2);   // current buffer complete
      } else {
        __builtin_amdgcn_s_wait_tensorcnt(0);
      }
    }
    __syncthreads();
#else
    {
      const uint4* gx = (const uint4*)(xb + (size_t)(mBlk + r) * BDIM + k0 + half * 16);
      const uint4* gh = (const uint4*)(hb + (size_t)(mBlk + r) * BDIM + k0 + half * 16);
      *(uint4*)(sx + cur * BUFE + r * LDSP + half * 16) = gx[0];
      *(uint4*)(sh + cur * BUFE + r * LDSP + half * 16) = gh[0];
    }
    __syncthreads();
#endif

    const __bf16* px = sx + cur * BUFE;
    const __bf16* ph = sh + cur * BUFE;

    Frag bx, bh;
    loadB(bx, wxa, ncol, k0, lane);
    loadB(bh, wha, ncol, k0, lane);
    if (k0 + 32 < BDIM) {
      __builtin_prefetch(wxa + (size_t)(ncol + (lane & 15)) * BDIM + k0 + 32, 0, 1);
      __builtin_prefetch(wha + (size_t)(ncol + (lane & 15)) * BDIM + k0 + 32, 0, 1);
    }

    Frag ax[4], ah[4];
#pragma unroll
    for (int j = 0; j < 4; ++j) {
      loadA(ax[j], px + (mq * 64 + j * 16) * LDSP, lane);
      loadA(ah[j], ph + (mq * 64 + j * 16) * LDSP, lane);
    }
#pragma unroll
    for (int j = 0; j < 4; ++j) {
      acc[j].v = __builtin_amdgcn_wmma_f32_16x16x32_bf16(
          false, ax[j].v, false, bx.v, (short)0, acc[j].v, false, false);
      acc[j].v = __builtin_amdgcn_wmma_f32_16x16x32_bf16(
          false, ah[j].v, false, bh.v, (short)0, acc[j].v, false, false);
    }
    __syncthreads();
  }

  const int   col  = ncol + (lane & 15);
  const float bsum = bxa[col] + bha[col];
#pragma unroll
  for (int j = 0; j < 4; ++j) {
    const int rbase = mBlk + mq * 64 + j * 16 + ((lane >> 4) << 3);
#pragma unroll
    for (int v = 0; v < 8; ++v) {
      const int   row = rbase + v;
      const float att = sigmoid_f(acc[j].f[v] + bsum);
      const float xv  = x[(size_t)row * BDIM + col];
      xg[(size_t)row * BDIM + col] = (__bf16)(att * xv);
    }
  }
}

// ---------------------------------------------------------------------------
// Phase 2: workgroup tile 64(M) x 64(N of hy); fully fused GRU.
// Per wave: 2 M-tiles x 3 gates x {input,hidden} = 12 fp32 accumulators.
__global__ __launch_bounds__(256) void phase2_kernel(
    const __bf16* __restrict__ xgb, const __bf16* __restrict__ hb,
    const __bf16* __restrict__ wih, const __bf16* __restrict__ whh,
    const float*  __restrict__ bih, const float*  __restrict__ bhh,
    const float*  __restrict__ hx,  float* __restrict__ hy)
{
  __shared__ __bf16 sa[2 * 64 * LDSP];
  __shared__ __bf16 sh[2 * 64 * LDSP];
  const int tid  = threadIdx.x;
  const int wave = tid >> 5;
  const int lane = tid & 31;
  const int mq   = wave >> 2;                 // 0..1 -> M offset mq*32
  const int nt   = wave & 3;
  const int mBlk = blockIdx.y * 64;
  const int nBlk = blockIdx.x * 64;
  const int ncol = nBlk + nt * 16;
  const int BUFE = 64 * LDSP;

  Acc ai[3][2], ah_[3][2];
#pragma unroll
  for (int g = 0; g < 3; ++g)
#pragma unroll
    for (int j = 0; j < 2; ++j)
#pragma unroll
      for (int v = 0; v < 8; ++v) { ai[g][j].f[v] = 0.0f; ah_[g][j].f[v] = 0.0f; }

#if USE_TDM
  const unsigned sa_lds = (unsigned)(size_t)(void*)&sa[0];
  const unsigned sh_lds = (unsigned)(size_t)(void*)&sh[0];
  if (wave == 0) {
    tdm_load_tile(sa_lds, xgb + (size_t)mBlk * BDIM, 64);
    tdm_load_tile(sh_lds, hb  + (size_t)mBlk * BDIM, 64);
  }
#else
  const int r = tid >> 2;                     // 0..63
  const int q = tid & 3;                      // 16B chunk of a 64B K-row
#endif

  for (int k0 = 0; k0 < BDIM; k0 += 32) {
    const int cur = (k0 >> 5) & 1;
#if USE_TDM
    if (wave == 0) {
      if (k0 + 32 < BDIM) {
        const int nxt = cur ^ 1;
        tdm_load_tile(sa_lds + (unsigned)(nxt * BUFE * 2),
                      xgb + (size_t)mBlk * BDIM + k0 + 32, 64);
        tdm_load_tile(sh_lds + (unsigned)(nxt * BUFE * 2),
                      hb + (size_t)mBlk * BDIM + k0 + 32, 64);
        __builtin_amdgcn_s_wait_tensorcnt(2);
      } else {
        __builtin_amdgcn_s_wait_tensorcnt(0);
      }
    }
    __syncthreads();
#else
    {
      const uint4* ga = (const uint4*)(xgb + (size_t)(mBlk + r) * BDIM + k0 + q * 8);
      const uint4* gh = (const uint4*)(hb  + (size_t)(mBlk + r) * BDIM + k0 + q * 8);
      *(uint4*)(sa + cur * BUFE + r * LDSP + q * 8) = ga[0];
      *(uint4*)(sh + cur * BUFE + r * LDSP + q * 8) = gh[0];
    }
    __syncthreads();
#endif

    // All 6 B fragments issued first so their L2 latency overlaps the ds loads.
    Frag bi[3], bh2[3];
#pragma unroll
    for (int g = 0; g < 3; ++g) {
      loadB(bi[g],  wih, g * BDIM + ncol, k0, lane);
      loadB(bh2[g], whh, g * BDIM + ncol, k0, lane);
      if (k0 + 32 < BDIM) {
        __builtin_prefetch(wih + (size_t)(g * BDIM + ncol + (lane & 15)) * BDIM + k0 + 32, 0, 1);
        __builtin_prefetch(whh + (size_t)(g * BDIM + ncol + (lane & 15)) * BDIM + k0 + 32, 0, 1);
      }
    }

    Frag fa[2], fh[2];
#pragma unroll
    for (int j = 0; j < 2; ++j) {
      loadA(fa[j], sa + cur * BUFE + (mq * 32 + j * 16) * LDSP, lane);
      loadA(fh[j], sh + cur * BUFE + (mq * 32 + j * 16) * LDSP, lane);
    }

#pragma unroll
    for (int g = 0; g < 3; ++g) {
#pragma unroll
      for (int j = 0; j < 2; ++j) {
        ai[g][j].v = __builtin_amdgcn_wmma_f32_16x16x32_bf16(
            false, fa[j].v, false, bi[g].v, (short)0, ai[g][j].v, false, false);
        ah_[g][j].v = __builtin_amdgcn_wmma_f32_16x16x32_bf16(
            false, fh[j].v, false, bh2[g].v, (short)0, ah_[g][j].v, false, false);
      }
    }
    __syncthreads();
  }

  const int   col  = ncol + (lane & 15);
  const float br_i = bih[col],            br_h = bhh[col];
  const float bz_i = bih[BDIM + col],     bz_h = bhh[BDIM + col];
  const float bn_i = bih[2 * BDIM + col], bn_h = bhh[2 * BDIM + col];
#pragma unroll
  for (int j = 0; j < 2; ++j) {
    const int rbase = mBlk + mq * 32 + j * 16 + ((lane >> 4) << 3);
#pragma unroll
    for (int v = 0; v < 8; ++v) {
      const int   row = rbase + v;
      const float rg = sigmoid_f(ai[0][j].f[v] + br_i + ah_[0][j].f[v] + br_h);
      const float zg = sigmoid_f(ai[1][j].f[v] + bz_i + ah_[1][j].f[v] + bz_h);
      const float ng = tanh_f(ai[2][j].f[v] + bn_i + rg * (ah_[2][j].f[v] + bn_h));
      const float hp = hx[(size_t)row * BDIM + col];
      hy[(size_t)row * BDIM + col] = (1.0f - zg) * ng + zg * hp;
    }
  }
}

// ---------------------------------------------------------------------------
extern "C" void kernel_launch(void* const* d_in, const int* in_sizes, int n_in,
                              void* d_out, int out_size, void* d_ws, size_t ws_size,
                              hipStream_t stream) {
  (void)in_sizes; (void)n_in; (void)out_size; (void)ws_size;

  const float* x   = (const float*)d_in[0];
  const float* hx  = (const float*)d_in[1];
  const float* Wxa = (const float*)d_in[2];
  const float* bxa = (const float*)d_in[3];
  const float* Wha = (const float*)d_in[4];
  const float* bha = (const float*)d_in[5];
  const float* Wih = (const float*)d_in[6];
  const float* bih = (const float*)d_in[7];
  const float* Whh = (const float*)d_in[8];
  const float* bhh = (const float*)d_in[9];
  float* hy = (float*)d_out;

  // Workspace (bf16): xb 16MB | hb 16MB | Wxa 2MB | Wha 2MB | Wih 6MB |
  // Whh 6MB | xg 16MB => 64 MB total.
  char* ws = (char*)d_ws;
  const size_t NBI = (size_t)BATCH * BDIM;              // 8Mi elements
  const size_t WSQ = (size_t)BDIM * BDIM;               // 1Mi elements
  __bf16* xb  = (__bf16*)(ws);
  __bf16* hb  = (__bf16*)(ws + NBI * 2);
  __bf16* wxa = (__bf16*)(ws + NBI * 4);
  __bf16* wha = (__bf16*)(ws + NBI * 4 + WSQ * 2);
  __bf16* wih = (__bf16*)(ws + NBI * 4 + WSQ * 4);
  __bf16* whh = (__bf16*)(ws + NBI * 4 + WSQ * 10);
  __bf16* xgb = (__bf16*)(ws + NBI * 4 + WSQ * 16);

  auto cvt = [&](const float* in, __bf16* out, size_t n) {
    const size_t per = 256 * 8;
    cvt_bf16_kernel<<<(unsigned)((n + per - 1) / per), 256, 0, stream>>>(in, out, (int)n);
  };
  cvt(x,   xb,  NBI);
  cvt(hx,  hb,  NBI);
  cvt(Wxa, wxa, WSQ);
  cvt(Wha, wha, WSQ);
  cvt(Wih, wih, 3 * WSQ);
  cvt(Whh, whh, 3 * WSQ);

  dim3 g1(BDIM / 64, BATCH / 128);   // (16, 64)
  phase1_kernel<<<g1, dim3(256), 0, stream>>>(xb, hb, wxa, wha, bxa, bha, x, xgb);

  dim3 g2(BDIM / 64, BATCH / 64);    // (16, 128)
  phase2_kernel<<<g2, dim3(256), 0, stream>>>(xgb, hb, wih, whh, bih, bhh, hx, hy);
}
